// Pointnet2_coarse_47863115546833
// MI455X (gfx1250) — matrix operand order repaired and checked by reading, not compiled, not run on hardware
//
#include <hip/hip_runtime.h>
#include <hip/hip_bf16.h>

// ---------------------------------------------------------------------------
// PointNet++ (coarse) forward for gfx1250 (MI455X).
// All GEMMs run on v_wmma_f32_16x16x32_bf16 with fragment-tiled operands:
//   X tiled: [B][C/32][N/16][32 lanes][16 bf16]  (one 32B load per lane)
//   A tiled: [O/16][C/32][32 lanes][16 bf16]     (contiguous k = 16*(lane/16) + elem)
// so the GEMM inner loop is pure global_load_b128 + v_wmma.
// ---------------------------------------------------------------------------

#define BATCH 4
#define BN_SCALE_C 0.99999500003749968f

typedef __bf16 v16bf __attribute__((ext_vector_type(16)));
typedef float  v8f   __attribute__((ext_vector_type(8)));
typedef unsigned int v4u __attribute__((ext_vector_type(4)));

__device__ __forceinline__ unsigned short f32_to_bf16(float f) {
  unsigned int x = __float_as_uint(f);
  unsigned int r = x + 0x7FFFu + ((x >> 16) & 1u);   // round-to-nearest-even
  return (unsigned short)(r >> 16);
}
__device__ __forceinline__ float bf16_to_f32(unsigned short h) {
  return __uint_as_float(((unsigned int)h) << 16);
}

// Element offset (within one batch) of logical (channel c, column n) in the
// B-operand tiled layout. Nt = N/16.
__device__ __forceinline__ size_t xtiled_off(int c, int nt, int nn, int Nt) {
  int kt   = c >> 5;
  int lane = nn + ((c >> 4) & 1) * 16;
  int p    = c & 15;
  return ((size_t)kt * Nt + nt) * 512 + (size_t)lane * 16 + p;
}

// ---------------------------------------------------------------------------
// Farthest point sampling: one block per batch, block-wide argmax with
// first-index tie-break (matches jnp.argmax).
// ---------------------------------------------------------------------------
__global__ void fps_kernel(const float* __restrict__ xyz, int N, int npoint,
                           int* __restrict__ idx_out, float* __restrict__ dist) {
  const int b   = blockIdx.x;
  const int tid = threadIdx.x;
  const int nt  = blockDim.x;
  const float* P = xyz + (size_t)b * N * 3;
  float* D = dist + (size_t)b * N;

  __shared__ float sval[256];
  __shared__ int   sidx[256];
  __shared__ int   slast;
  if (tid == 0) slast = 0;
  __syncthreads();

  for (int it = 0; it < npoint; ++it) {
    int last = slast;
    if (tid == 0) idx_out[(size_t)b * npoint + it] = last;
    float lx = P[last * 3 + 0], ly = P[last * 3 + 1], lz = P[last * 3 + 2];
    float bestv = -1.0f; int besti = 0;
    for (int i = tid; i < N; i += nt) {
      float dx = P[i * 3 + 0] - lx;
      float dy = P[i * 3 + 1] - ly;
      float dz = P[i * 3 + 2] - lz;
      float d  = dx * dx + dy * dy + dz * dz;
      float dd = (it == 0) ? d : fminf(D[i], d);
      D[i] = dd;
      if (dd > bestv) { bestv = dd; besti = i; }
    }
    sval[tid] = bestv; sidx[tid] = besti;
    __syncthreads();
    for (int s = nt >> 1; s > 0; s >>= 1) {
      if (tid < s) {
        float ov = sval[tid + s]; int oi = sidx[tid + s];
        if (ov > sval[tid] || (ov == sval[tid] && oi < sidx[tid])) {
          sval[tid] = ov; sidx[tid] = oi;
        }
      }
      __syncthreads();
    }
    if (tid == 0) slast = sidx[0];
    __syncthreads();
  }
}

__global__ void gather_xyz_kernel(const float* __restrict__ xyz,
                                  const int* __restrict__ idx,
                                  int N, int S, float* __restrict__ out) {
  int t = blockIdx.x * blockDim.x + threadIdx.x;
  if (t >= BATCH * S) return;
  int b = t / S, s = t % S;
  int gi = idx[(size_t)b * S + s];
  const float* p = xyz + ((size_t)b * N + gi) * 3;
  float* o = out + ((size_t)b * S + s) * 3;
  o[0] = p[0]; o[1] = p[1]; o[2] = p[2];
}

// First K in-radius indices (== reference top_k(-score)); pad with first hit.
__global__ void ball_query_kernel(const float* __restrict__ xyz,
                                  const float* __restrict__ new_xyz,
                                  int N, int S, float r2, int K,
                                  int* __restrict__ gidx) {
  int t = blockIdx.x * blockDim.x + threadIdx.x;
  if (t >= BATCH * S) return;
  int b = t / S, s = t % S;
  const float* P = xyz + (size_t)b * N * 3;
  const float* Q = new_xyz + ((size_t)b * S + s) * 3;
  float qx = Q[0], qy = Q[1], qz = Q[2];
  int* out = gidx + ((size_t)b * S + s) * K;
  int cnt = 0, first = 0;
  for (int i = 0; i < N && cnt < K; ++i) {
    float dx = P[i * 3 + 0] - qx;
    float dy = P[i * 3 + 1] - qy;
    float dz = P[i * 3 + 2] - qz;
    if (dx * dx + dy * dy + dz * dz < r2) {
      if (cnt == 0) first = i;
      out[cnt++] = i;
    }
  }
  for (; cnt < K; ++cnt) out[cnt] = first;
}

// grouped (tiled bf16, Cpad channels): c<3 centered xyz; 3..3+Cf feats; pad 0.
__global__ void group_kernel(const float* __restrict__ xyz,
                             const float* __restrict__ new_xyz,
                             const float* __restrict__ feats,
                             const int* __restrict__ gidx,
                             int N, int S, int K, int Cf, int Cpad,
                             unsigned short* __restrict__ grouped) {
  int t = blockIdx.x * blockDim.x + threadIdx.x;
  int total = BATCH * S * K;
  if (t >= total) return;
  int b = t / (S * K), rem = t % (S * K), s = rem / K, k = rem % K;
  int gi = gidx[((size_t)b * S + s) * K + k];
  const float* P = xyz + ((size_t)b * N + gi) * 3;
  const float* Q = new_xyz + ((size_t)b * S + s) * 3;
  int n  = s * K + k;
  int Nt = (S * K) >> 4, nt = n >> 4, nn = n & 15;
  unsigned short* G = grouped + (size_t)b * (Cpad >> 5) * Nt * 512;
  G[xtiled_off(0, nt, nn, Nt)] = f32_to_bf16(P[0] - Q[0]);
  G[xtiled_off(1, nt, nn, Nt)] = f32_to_bf16(P[1] - Q[1]);
  G[xtiled_off(2, nt, nn, Nt)] = f32_to_bf16(P[2] - Q[2]);
  for (int c = 0; c < Cf; ++c)
    G[xtiled_off(3 + c, nt, nn, Nt)] =
        f32_to_bf16(feats[((size_t)b * Cf + c) * N + gi]);
  for (int c = 3 + Cf; c < Cpad; ++c) G[xtiled_off(c, nt, nn, Nt)] = 0;
}

// sa_global grouping: tiled (B, 1056, 128) = concat(xyz4^T, f4), padded.
__global__ void sag_group_kernel(const float* __restrict__ xyz4,
                                 const float* __restrict__ f4,
                                 unsigned short* __restrict__ grouped) {
  int t = blockIdx.x * blockDim.x + threadIdx.x;
  if (t >= BATCH * 128) return;
  int b = t / 128, k = t % 128;
  int Nt = 8, nt = k >> 4, nn = k & 15;
  unsigned short* G = grouped + (size_t)b * 33 * Nt * 512;  // 1056/32 = 33
  G[xtiled_off(0, nt, nn, Nt)] = f32_to_bf16(xyz4[((size_t)b * 128 + k) * 3 + 0]);
  G[xtiled_off(1, nt, nn, Nt)] = f32_to_bf16(xyz4[((size_t)b * 128 + k) * 3 + 1]);
  G[xtiled_off(2, nt, nn, Nt)] = f32_to_bf16(xyz4[((size_t)b * 128 + k) * 3 + 2]);
  for (int c = 0; c < 1024; ++c)
    G[xtiled_off(3 + c, nt, nn, Nt)] =
        f32_to_bf16(f4[((size_t)b * 1024 + c) * 128 + k]);
  for (int c = 1027; c < 1056; ++c) G[xtiled_off(c, nt, nn, Nt)] = 0;
}

// Weights f32 (O x C) -> A-fragment-tiled bf16 [O/16][Cpad/32][32][16].
__global__ void wconv_kernel(const float* __restrict__ W, int O, int C, int Cpad,
                             unsigned short* __restrict__ Wp) {
  int t = blockIdx.x * blockDim.x + threadIdx.x;
  if (t >= O * Cpad) return;
  int o = t / Cpad, c = t % Cpad;
  int mt = o >> 4, m = o & 15, kt = c >> 5, kk = c & 31;
  int lane, p;
  lane = m + (kk >> 4) * 16; p = kk & 15;
  int Ck = Cpad >> 5;
  size_t off = (((size_t)mt * Ck + kt) * 32 + lane) * 16 + p;
  Wp[off] = (c < C) ? f32_to_bf16(W[(size_t)o * C + c]) : (unsigned short)0;
}

// ---------------------------------------------------------------------------
// WMMA GEMM + BN + ReLU. One wave per 16(M)x64(N) strip: per 32-K step,
// 1 A-frag load (32B) + 4 B-frag loads (32B) + 4 v_wmma_f32_16x16x32_bf16.
// Writes next layer's tiled bf16 (16B packed stores) and/or row-major f32.
// ---------------------------------------------------------------------------
__global__ __launch_bounds__(32)
void wmma_gemm_bn_relu(const unsigned short* __restrict__ At,
                       const unsigned short* __restrict__ Xt,
                       const float* __restrict__ gamma,
                       const float* __restrict__ beta,
                       int O, int C, int N,
                       unsigned short* __restrict__ outB,   // tiled bf16 or null
                       float* __restrict__ outF) {          // row-major f32 or null
  const int lane   = threadIdx.x;
  const int nt0    = blockIdx.x * 4;
  const int mt     = blockIdx.y;
  const int b      = blockIdx.z;
  const int Ck     = C >> 5;
  const int Nt     = N >> 4;

  const unsigned short* Ab = At + (size_t)mt * Ck * 512 + (size_t)lane * 16;
  const unsigned short* Xb = Xt + (size_t)b * Ck * Nt * 512 + (size_t)lane * 16;

  v8f acc0 = {}, acc1 = {}, acc2 = {}, acc3 = {};
  for (int kt = 0; kt < Ck; ++kt) {
    v16bf a = *(const v16bf*)(Ab + (size_t)kt * 512);
    const unsigned short* Xk = Xb + ((size_t)kt * Nt + nt0) * 512;
    v16bf b0 = *(const v16bf*)(Xk);
    v16bf b1 = *(const v16bf*)(Xk + 512);
    v16bf b2 = *(const v16bf*)(Xk + 1024);
    v16bf b3 = *(const v16bf*)(Xk + 1536);
    if (kt + 1 < Ck) {
      __builtin_prefetch(Xb + ((size_t)(kt + 1) * Nt + nt0) * 512, 0, 3);
      __builtin_prefetch(Ab + (size_t)(kt + 1) * 512, 0, 3);
    }
    acc0 = __builtin_amdgcn_wmma_f32_16x16x32_bf16(false, a, false, b0, (short)0, acc0, false, false);
    acc1 = __builtin_amdgcn_wmma_f32_16x16x32_bf16(false, a, false, b1, (short)0, acc1, false, false);
    acc2 = __builtin_amdgcn_wmma_f32_16x16x32_bf16(false, a, false, b2, (short)0, acc2, false, false);
    acc3 = __builtin_amdgcn_wmma_f32_16x16x32_bf16(false, a, false, b3, (short)0, acc3, false, false);
  }

  // D layout: vgpr r -> M = r (lanes 0-15), r+8 (lanes 16-31); N = lane&15.
  const int mhalf = lane >> 4;
  const int p0    = mhalf * 8;
  const int nn    = lane & 15;
  float g[8], bb[8];
#pragma unroll
  for (int r = 0; r < 8; ++r) {
    int m = mt * 16 + p0 + r;
    g[r]  = gamma[m];
    bb[r] = beta[m];
  }
  v8f accs[4] = {acc0, acc1, acc2, acc3};
  // Next-layer tiled coords (constant per block): kt2 = mt/2, lane2 from mt&1.
  const int kt2   = mt >> 1;
  const int lane2 = nn + (mt & 1) * 16;
  unsigned short* OB =
      outB ? outB + ((size_t)b * (O >> 5)) * Nt * 512 : (unsigned short*)0;
#pragma unroll
  for (int t = 0; t < 4; ++t) {
    int nt = nt0 + t;
    float v[8];
#pragma unroll
    for (int r = 0; r < 8; ++r) {
      float x = g[r] * accs[t][r] * BN_SCALE_C + bb[r];
      v[r] = x > 0.0f ? x : 0.0f;
    }
    if (OB) {
      union { unsigned short s[8]; v4u u; } pk;
#pragma unroll
      for (int r = 0; r < 8; ++r) pk.s[r] = f32_to_bf16(v[r]);
      *(v4u*)(OB + ((size_t)kt2 * Nt + nt) * 512 + (size_t)lane2 * 16 + p0) = pk.u;
    }
    if (outF) {
#pragma unroll
      for (int r = 0; r < 8; ++r) {
        int m = mt * 16 + p0 + r;
        outF[((size_t)b * O + m) * N + nt * 16 + nn] = v[r];
      }
    }
  }
}

// max over K axis, reading the tiled bf16 activations -> f32 (B, O, S)
__global__ void maxpool_tiled_kernel(const unsigned short* __restrict__ Xt,
                                     int O, int S, int K, float* __restrict__ out) {
  int t = blockIdx.x * blockDim.x + threadIdx.x;
  int total = BATCH * O * S;
  if (t >= total) return;
  int b = t / (O * S), rem = t % (O * S), o = rem / S, s = rem % S;
  int Nt = (S * K) >> 4;
  const unsigned short* G = Xt + (size_t)b * (O >> 5) * Nt * 512;
  float m = -3.4e38f;
  for (int k = 0; k < K; ++k) {
    int n = s * K + k;
    m = fmaxf(m, bf16_to_f32(G[xtiled_off(o, n >> 4, n & 15, Nt)]));
  }
  out[((size_t)b * O + o) * S + s] = m;
}

// 3-NN (keeps first-occurrence order like top_k) + inverse-distance weights.
__global__ void fp_3nn_kernel(const float* __restrict__ unknown,
                              const float* __restrict__ known,
                              int Nu, int Nk,
                              int* __restrict__ nnidx, float* __restrict__ nnw) {
  int t = blockIdx.x * blockDim.x + threadIdx.x;
  if (t >= BATCH * Nu) return;
  int b = t / Nu, n = t % Nu;
  const float* q = unknown + ((size_t)b * Nu + n) * 3;
  const float* P = known + (size_t)b * Nk * 3;
  float qx = q[0], qy = q[1], qz = q[2];
  float d0 = 3.4e38f, d1 = 3.4e38f, d2 = 3.4e38f;
  int   i0 = 0, i1 = 0, i2 = 0;
  for (int i = 0; i < Nk; ++i) {
    float dx = P[i * 3 + 0] - qx;
    float dy = P[i * 3 + 1] - qy;
    float dz = P[i * 3 + 2] - qz;
    float d = dx * dx + dy * dy + dz * dz;
    if (d < d0)      { d2 = d1; i2 = i1; d1 = d0; i1 = i0; d0 = d; i0 = i; }
    else if (d < d1) { d2 = d1; i2 = i1; d1 = d;  i1 = i; }
    else if (d < d2) { d2 = d;  i2 = i; }
  }
  float r0 = 1.0f / (d0 + 1e-8f), r1 = 1.0f / (d1 + 1e-8f), r2 = 1.0f / (d2 + 1e-8f);
  float s = r0 + r1 + r2;
  size_t base = ((size_t)b * Nu + n) * 3;
  nnidx[base + 0] = i0; nnidx[base + 1] = i1; nnidx[base + 2] = i2;
  nnw[base + 0] = r0 / s; nnw[base + 1] = r1 / s; nnw[base + 2] = r2 / s;
}

// concat(interp(known_feats), unknown_feats) -> tiled bf16 (B, 2*Cf, Nu)
__global__ void fp_concat_kernel(const float* __restrict__ kf,  // (B,Cf,Nk)
                                 const float* __restrict__ uf,  // (B,Cf,Nu)
                                 const int* __restrict__ nnidx,
                                 const float* __restrict__ nnw,
                                 int Cf, int Nu, int Nk,
                                 unsigned short* __restrict__ cat) {
  int t = blockIdx.x * blockDim.x + threadIdx.x;
  if (t >= BATCH * Nu) return;
  int b = t / Nu, n = t % Nu;
  size_t base3 = ((size_t)b * Nu + n) * 3;
  int   i0 = nnidx[base3 + 0], i1 = nnidx[base3 + 1], i2 = nnidx[base3 + 2];
  float w0 = nnw[base3 + 0],  w1 = nnw[base3 + 1],  w2 = nnw[base3 + 2];
  int Nt = Nu >> 4, nt = n >> 4, nn = n & 15;
  unsigned short* G = cat + (size_t)b * ((2 * Cf) >> 5) * Nt * 512;
  for (int c = 0; c < Cf; ++c) {
    const float* kr = kf + ((size_t)b * Cf + c) * Nk;
    float v = w0 * kr[i0] + w1 * kr[i1] + w2 * kr[i2];
    G[xtiled_off(c, nt, nn, Nt)] = f32_to_bf16(v);
    G[xtiled_off(Cf + c, nt, nn, Nt)] =
        f32_to_bf16(uf[((size_t)b * Cf + c) * Nu + n]);
  }
}

// ---------------------------------------------------------------------------
// Host orchestration
// ---------------------------------------------------------------------------
static inline int cdiv(int a, int b) { return (a + b - 1) / b; }

static void run_layer(const float* Wf, const float* g, const float* bt,
                      int O, int C, int Cpad, int N,
                      const unsigned short* in, unsigned short* outB, float* outF,
                      unsigned short* wpad, hipStream_t stream) {
  int total = O * Cpad;
  wconv_kernel<<<cdiv(total, 256), 256, 0, stream>>>(Wf, O, C, Cpad, wpad);
  dim3 grid(N / 64, O / 16, BATCH);
  wmma_gemm_bn_relu<<<grid, 32, 0, stream>>>(wpad, in, g, bt, O, Cpad, N, outB, outF);
}

extern "C" void kernel_launch(void* const* d_in, const int* in_sizes, int n_in,
                              void* d_out, int out_size, void* d_ws, size_t ws_size,
                              hipStream_t stream) {
  (void)in_sizes; (void)n_in; (void)out_size; (void)ws_size;
  const float* pc = (const float*)d_in[0];   // (4, 16384, 3)

  const int SA1 = 1, SA2 = 10, SA3 = 19, SA4 = 28, SAG = 37, FP1 = 46;
  auto W  = [&](int base, int l) { return (const float*)d_in[base + 3 * l + 0]; };
  auto G  = [&](int base, int l) { return (const float*)d_in[base + 3 * l + 1]; };
  auto Bb = [&](int base, int l) { return (const float*)d_in[base + 3 * l + 2]; };

  char* ws = (char*)d_ws;
  size_t off = 0;
  auto alloc = [&](size_t bytes) -> void* {
    void* p = (void*)(ws + off);
    off += (bytes + 255) & ~(size_t)255;
    return p;
  };
  float* dist   = (float*)alloc((size_t)BATCH * 16384 * 4);
  int*   fpsidx = (int*)  alloc((size_t)BATCH * 1024 * 4);
  float* xyz1   = (float*)alloc((size_t)BATCH * 1024 * 3 * 4);
  float* xyz2   = (float*)alloc((size_t)BATCH * 384 * 3 * 4);
  float* xyz3   = (float*)alloc((size_t)BATCH * 256 * 3 * 4);
  float* xyz4   = (float*)alloc((size_t)BATCH * 128 * 3 * 4);
  int*   gidx   = (int*)  alloc((size_t)BATCH * 1024 * 64 * 4);
  int*   nnidx  = (int*)  alloc((size_t)BATCH * 384 * 3 * 4);
  float* nnw    = (float*)alloc((size_t)BATCH * 384 * 3 * 4);
  float* f1     = (float*)alloc((size_t)BATCH * 128 * 1024 * 4);
  float* f2     = (float*)alloc((size_t)BATCH * 1024 * 384 * 4);
  float* f3     = (float*)alloc((size_t)BATCH * 1024 * 256 * 4);
  float* f4     = (float*)alloc((size_t)BATCH * 1024 * 128 * 4);
  float* ffp    = (float*)alloc((size_t)BATCH * 1024 * 256 * 4);
  unsigned short* wpad = (unsigned short*)alloc((size_t)1024 * 2048 * 2);
  unsigned short* P = (unsigned short*)alloc((size_t)25165824 * 2);
  unsigned short* Q = (unsigned short*)alloc((size_t)50331648 * 2);

  float* gfeat  = (float*)d_out;                 // (4,1024,1)
  float* coarse = (float*)d_out + BATCH * 1024;  // (4,1024,384)

  // ---------------- SA1: N=16384 -> S=1024, K=64, C 3->32/64/128 ----------
  fps_kernel<<<BATCH, 256, 0, stream>>>(pc, 16384, 1024, fpsidx, dist);
  gather_xyz_kernel<<<cdiv(BATCH * 1024, 256), 256, 0, stream>>>(pc, fpsidx, 16384, 1024, xyz1);
  ball_query_kernel<<<cdiv(BATCH * 1024, 256), 256, 0, stream>>>(pc, xyz1, 16384, 1024, 0.0016f, 64, gidx);
  group_kernel<<<cdiv(BATCH * 1024 * 64, 256), 256, 0, stream>>>(pc, xyz1, nullptr, gidx, 16384, 1024, 64, 0, 32, P);
  run_layer(W(SA1,0), G(SA1,0), Bb(SA1,0),  32,  3, 32, 65536, P, Q, nullptr, wpad, stream);
  run_layer(W(SA1,1), G(SA1,1), Bb(SA1,1),  64, 32, 32, 65536, Q, P, nullptr, wpad, stream);
  run_layer(W(SA1,2), G(SA1,2), Bb(SA1,2), 128, 64, 64, 65536, P, Q, nullptr, wpad, stream);
  maxpool_tiled_kernel<<<cdiv(BATCH * 128 * 1024, 256), 256, 0, stream>>>(Q, 128, 1024, 64, f1);

  // ---------------- SA2: 1024 -> 384, K=32, C 131->256/512/1024 -----------
  fps_kernel<<<BATCH, 256, 0, stream>>>(xyz1, 1024, 384, fpsidx, dist);
  gather_xyz_kernel<<<cdiv(BATCH * 384, 256), 256, 0, stream>>>(xyz1, fpsidx, 1024, 384, xyz2);
  ball_query_kernel<<<cdiv(BATCH * 384, 256), 256, 0, stream>>>(xyz1, xyz2, 1024, 384, 0.0064f, 32, gidx);
  group_kernel<<<cdiv(BATCH * 384 * 32, 256), 256, 0, stream>>>(xyz1, xyz2, f1, gidx, 1024, 384, 32, 128, 160, P);
  run_layer(W(SA2,0), G(SA2,0), Bb(SA2,0),  256, 131, 160, 12288, P, Q, nullptr, wpad, stream);
  run_layer(W(SA2,1), G(SA2,1), Bb(SA2,1),  512, 256, 256, 12288, Q, P, nullptr, wpad, stream);
  run_layer(W(SA2,2), G(SA2,2), Bb(SA2,2), 1024, 512, 512, 12288, P, Q, nullptr, wpad, stream);
  maxpool_tiled_kernel<<<cdiv(BATCH * 1024 * 384, 256), 256, 0, stream>>>(Q, 1024, 384, 32, f2);

  // ---------------- SA3: 384 -> 256, K=16, C 1027->512/512/1024 -----------
  fps_kernel<<<BATCH, 256, 0, stream>>>(xyz2, 384, 256, fpsidx, dist);
  gather_xyz_kernel<<<cdiv(BATCH * 256, 256), 256, 0, stream>>>(xyz2, fpsidx, 384, 256, xyz3);
  ball_query_kernel<<<cdiv(BATCH * 256, 256), 256, 0, stream>>>(xyz2, xyz3, 384, 256, 0.0256f, 16, gidx);
  group_kernel<<<cdiv(BATCH * 256 * 16, 256), 256, 0, stream>>>(xyz2, xyz3, f2, gidx, 384, 256, 16, 1024, 1056, P);
  run_layer(W(SA3,0), G(SA3,0), Bb(SA3,0),  512, 1027, 1056, 4096, P, Q, nullptr, wpad, stream);
  run_layer(W(SA3,1), G(SA3,1), Bb(SA3,1),  512,  512,  512, 4096, Q, P, nullptr, wpad, stream);
  run_layer(W(SA3,2), G(SA3,2), Bb(SA3,2), 1024,  512,  512, 4096, P, Q, nullptr, wpad, stream);
  maxpool_tiled_kernel<<<cdiv(BATCH * 1024 * 256, 256), 256, 0, stream>>>(Q, 1024, 256, 16, f3);

  // ---------------- SA4: 256 -> 128, K=8, C 1027->512/512/1024 ------------
  fps_kernel<<<BATCH, 256, 0, stream>>>(xyz3, 256, 128, fpsidx, dist);
  gather_xyz_kernel<<<cdiv(BATCH * 128, 256), 256, 0, stream>>>(xyz3, fpsidx, 256, 128, xyz4);
  ball_query_kernel<<<cdiv(BATCH * 128, 256), 256, 0, stream>>>(xyz3, xyz4, 256, 128, 0.1024f, 8, gidx);
  group_kernel<<<cdiv(BATCH * 128 * 8, 256), 256, 0, stream>>>(xyz3, xyz4, f3, gidx, 256, 128, 8, 1024, 1056, P);
  run_layer(W(SA4,0), G(SA4,0), Bb(SA4,0),  512, 1027, 1056, 1024, P, Q, nullptr, wpad, stream);
  run_layer(W(SA4,1), G(SA4,1), Bb(SA4,1),  512,  512,  512, 1024, Q, P, nullptr, wpad, stream);
  run_layer(W(SA4,2), G(SA4,2), Bb(SA4,2), 1024,  512,  512, 1024, P, Q, nullptr, wpad, stream);
  maxpool_tiled_kernel<<<cdiv(BATCH * 1024 * 128, 256), 256, 0, stream>>>(Q, 1024, 128, 8, f4);

  // ---------------- SA global: (1027,1,128) -> gfeat (B,1024,1) -----------
  sag_group_kernel<<<cdiv(BATCH * 128, 128), 128, 0, stream>>>(xyz4, f4, P);
  run_layer(W(SAG,0), G(SAG,0), Bb(SAG,0),  512, 1027, 1056, 128, P, Q, nullptr, wpad, stream);
  run_layer(W(SAG,1), G(SAG,1), Bb(SAG,1),  512,  512,  512, 128, Q, P, nullptr, wpad, stream);
  run_layer(W(SAG,2), G(SAG,2), Bb(SAG,2), 1024,  512,  512, 128, P, Q, nullptr, wpad, stream);
  maxpool_tiled_kernel<<<cdiv(BATCH * 1024, 256), 256, 0, stream>>>(Q, 1024, 1, 128, gfeat);

  // ---------------- FP a: (xyz3, xyz4, f3, f4) -> ffp (B,1024,256) --------
  fp_3nn_kernel<<<cdiv(BATCH * 256, 256), 256, 0, stream>>>(xyz3, xyz4, 256, 128, nnidx, nnw);
  fp_concat_kernel<<<cdiv(BATCH * 256, 256), 256, 0, stream>>>(f4, f3, nnidx, nnw, 1024, 256, 128, P);
  run_layer(W(FP1,0), G(FP1,0), Bb(FP1,0), 1024, 2048, 2048, 256, P, nullptr, ffp, wpad, stream);

  // ---------------- FP b: (xyz2, xyz3, f2, ffp) -> coarse (B,1024,384) ----
  fp_3nn_kernel<<<cdiv(BATCH * 384, 256), 256, 0, stream>>>(xyz2, xyz3, 384, 256, nnidx, nnw);
  fp_concat_kernel<<<cdiv(BATCH * 384, 256), 256, 0, stream>>>(ffp, f2, nnidx, nnw, 1024, 384, 256, P);
  run_layer(W(FP1,0), G(FP1,0), Bb(FP1,0), 1024, 2048, 2048, 384, P, nullptr, coarse, wpad, stream);
}
